// DeformConv2d_44117904065044
// MI455X (gfx1250) — matrix operand order repaired and checked by reading, not compiled, not run on hardware
//
#include <hip/hip_runtime.h>

typedef _Float16 v4h  __attribute__((ext_vector_type(4)));
typedef _Float16 v8h  __attribute__((ext_vector_type(8)));
typedef _Float16 v16h __attribute__((ext_vector_type(16)));
typedef float    v8f  __attribute__((ext_vector_type(8)));

#define B_       8
#define CIN      256
#define H_       64
#define W_       64
#define COUT     256
#define HW       4096            // H_*W_
#define KTOT     2304            // CIN * 9
#define KSTEPS   72              // KTOT / 32
#define NTILE    64
#define WP_ELEMS (KTOT * COUT)   // 589824 f16
#define WP_BYTES (WP_ELEMS * 2)  // 1179648 bytes (256B aligned)

#define AS1 __attribute__((address_space(1)))
#define AS3 __attribute__((address_space(3)))

#if __has_builtin(__builtin_amdgcn_global_load_async_to_lds_b32)
#define HAVE_ASYNC_LDS 1
#else
#define HAVE_ASYNC_LDS 0
#endif

__device__ __forceinline__ void wait_async0() {
#if __has_builtin(__builtin_amdgcn_s_wait_asynccnt)
  __builtin_amdgcn_s_wait_asynccnt(0);
#else
  asm volatile("s_wait_asynccnt 0x0" ::: "memory");
#endif
  asm volatile("" ::: "memory");  // keep LDS reads after the wait
}

// ---------------------------------------------------------------------------
// Pre-pass 1: pack weight (Cout,Cin,3,3) f32 -> f16 in exact per-lane WMMA
// A-fragment order: Wp[s][mtile(16)][lane(32)][j(16)], where for lane
// (hl = lane>>4, l15 = lane&15) element j holds
//   A[m = mtile*16 + l15][k = s*32 + (j&8)*2 + hl*8 + (j&7)]
// with K ordered as k = kid*256 + c (tap outer, channel inner).
// A wave then loads a whole v16h fragment as one contiguous 32B read.
// ---------------------------------------------------------------------------
__global__ __launch_bounds__(256) void pack_weight_kernel(
    const float* __restrict__ w, _Float16* __restrict__ wp) {
  const int idx = blockIdx.x * 256 + threadIdx.x;   // < 589824
  const int j   = idx & 15;
  const int ln  = (idx >> 4) & 31;
  const int mt  = (idx >> 9) & 15;
  const int s   = idx >> 13;
  const int hl  = ln >> 4;
  const int m   = mt * 16 + (ln & 15);
  const int k   = s * 32 + ((j & 8) * 2 + hl * 8 + (j & 7));
  const int kid = k >> 8;
  const int c   = k & 255;
  wp[idx] = (_Float16)w[(size_t)m * KTOT + c * 9 + kid];
}

// ---------------------------------------------------------------------------
// Pre-pass 2: x (B,C,H,W) f32 -> xT (B,H,W,C) f16 via LDS-tiled transpose.
// NHWC f16 (16.8 MB) keeps the bilinear gathers coalesced and L2-resident.
// ---------------------------------------------------------------------------
__global__ __launch_bounds__(256) void transpose_x_kernel(
    const float* __restrict__ x, _Float16* __restrict__ xt) {
  __shared__ float tile[32][33];
  const int b  = blockIdx.z;
  const int c0 = blockIdx.y * 32;
  const int s0 = blockIdx.x * 32;
  const int tx = threadIdx.x;   // 0..31
  const int ty = threadIdx.y;   // 0..7
  const float* src = x + ((size_t)b * CIN + c0) * HW + s0;
#pragma unroll
  for (int i = 0; i < 4; ++i)
    tile[ty + i * 8][tx] = src[(size_t)(ty + i * 8) * HW + tx];
  __syncthreads();
  _Float16* dst = xt + ((size_t)b * HW + s0) * CIN + c0;
#pragma unroll
  for (int i = 0; i < 4; ++i)
    dst[(size_t)(ty + i * 8) * CIN + tx] = (_Float16)tile[tx][ty + i * 8];
}

// ---------------------------------------------------------------------------
// Main fused kernel: implicit GEMM with WMMA.
//   D[m=cout][n=spatial] = sum_k Wp[m][k] * bilinear_sample[k][n] + bias[m]
// Workgroup: 256 threads (8 waves) owns 256 cout x 64 spatial; K-step = 32.
// B (sampled) tile double-buffered in LDS -> 1 barrier per K-step.
// A (weight) fragments read directly from global (no cross-wave reuse).
// ---------------------------------------------------------------------------
__global__ __launch_bounds__(256) void deform_wmma_kernel(
    const _Float16* __restrict__ xt, const _Float16* __restrict__ wp,
    const float* __restrict__ off, const float* __restrict__ bias,
    float* __restrict__ out) {
  // Rows padded to 40 f16 (80B): conflict-free ds_load_b128 fragment loads.
  __shared__ _Float16 Blds[2 * NTILE * 40];   // 10 KB, double-buffered
  __shared__ v4h  meta_wh[NTILE];             // bilinear weights (f16x4)
  __shared__ int4 meta_i[NTILE];              // clamped corner bases in xT
#if HAVE_ASYNC_LDS
  __shared__ float off_stage[2][NTILE];       // async-staged offset rows
#endif

  const int tid  = threadIdx.x;
  const int lane = tid & 31;
  const int wv   = tid >> 5;    // wave 0..7
  const int hl   = lane >> 4;   // half-wave
  const int l15  = lane & 15;

  const int n0  = blockIdx.x * NTILE;   // global spatial base
  const int b   = n0 >> 12;             // image (NTILE divides HW)
  const int sp0 = n0 & (HW - 1);

  v8f acc[2][4];
#pragma unroll
  for (int i = 0; i < 2; ++i)
#pragma unroll
    for (int j = 0; j < 4; ++j)
#pragma unroll
      for (int r = 0; r < 8; ++r) acc[i][j][r] = 0.0f;

  for (int s = 0; s < KSTEPS; ++s) {
    const int kid  = s >> 3;           // kernel tap 0..8
    const int cb   = (s & 7) << 5;     // channel block base
    const int bufo = (s & 1) * (NTILE * 40);

    // Refresh bilinear metadata when the kernel tap changes (9 of 72 steps).
    if ((s & 7) == 0) {
      if (tid < NTILE) {
        float dy, dx;
#if HAVE_ASYNC_LDS
        // CDNA5 async global->LDS staging of the two offset rows (ASYNCcnt).
        const float* gsrc = off + ((size_t)b * 18 + kid * 2) * HW + sp0 + tid;
        __builtin_amdgcn_global_load_async_to_lds_b32(
            (AS1 int*)(void*)gsrc, (AS3 int*)(void*)&off_stage[0][tid], 0, 0);
        __builtin_amdgcn_global_load_async_to_lds_b32(
            (AS1 int*)(void*)(gsrc + HW), (AS3 int*)(void*)&off_stage[1][tid],
            0, 0);
        wait_async0();
        dy = off_stage[0][tid];
        dx = off_stage[1][tid];
#else
        const float* ob = off + ((size_t)b * 18 + kid * 2) * HW + sp0 + tid;
        dy = ob[0];
        dx = ob[HW];
#endif
        const int sp = sp0 + tid;
        const int ho = sp >> 6, wo = sp & 63;
        const int kh = kid / 3, kw = kid - kh * 3;
        const float py = dy + (float)(ho - 1 + kh);
        const float px = dx + (float)(wo - 1 + kw);
        const float y0f = floorf(py);
        const float x0f = floorf(px);
        const float wy = py - y0f;
        const float wx = px - x0f;
        const int y0 = (int)y0f, x0 = (int)x0f;
        const int y1 = y0 + 1,  x1 = x0 + 1;
        const float vy0 = (y0 >= 0 && y0 < H_) ? 1.0f : 0.0f;
        const float vy1 = (y1 >= 0 && y1 < H_) ? 1.0f : 0.0f;
        const float vx0 = (x0 >= 0 && x0 < W_) ? 1.0f : 0.0f;
        const float vx1 = (x1 >= 0 && x1 < W_) ? 1.0f : 0.0f;
        const int cy0 = y0 < 0 ? 0 : (y0 > H_ - 1 ? H_ - 1 : y0);
        const int cy1 = y1 < 0 ? 0 : (y1 > H_ - 1 ? H_ - 1 : y1);
        const int cx0 = x0 < 0 ? 0 : (x0 > W_ - 1 ? W_ - 1 : x0);
        const int cx1 = x1 < 0 ? 0 : (x1 > W_ - 1 ? W_ - 1 : x1);
        v4h wh;
        wh.x = (_Float16)((1.0f - wy) * (1.0f - wx) * vy0 * vx0);
        wh.y = (_Float16)((1.0f - wy) * wx * vy0 * vx1);
        wh.z = (_Float16)(wy * (1.0f - wx) * vy1 * vx0);
        wh.w = (_Float16)(wy * wx * vy1 * vx1);
        meta_wh[tid] = wh;
        const int base = b * HW * CIN;
        meta_i[tid] = make_int4(base + (cy0 * W_ + cx0) * CIN,
                                base + (cy0 * W_ + cx1) * CIN,
                                base + (cy1 * W_ + cx0) * CIN,
                                base + (cy1 * W_ + cx1) * CIN);
      }
      __syncthreads();  // publish metadata
    }

    // Sample B tile into Blds[buf]: lane gathers v4h (8B, 4 channels) per
    // corner -> 64B coalesced per corner per 8-lane group; packed-f16 FMA.
#pragma unroll
    for (int q = 0; q < 2; ++q) {
      const int p  = wv * 8 + q * 4 + (lane >> 3);
      const int cl = (lane & 7) * 4;
      const v4h  w4 = meta_wh[p];
      const int4 ix = meta_i[p];
      const int c = cb + cl;
      const v4h s00 = *(const v4h*)(xt + ix.x + c);
      const v4h s01 = *(const v4h*)(xt + ix.y + c);
      const v4h s10 = *(const v4h*)(xt + ix.z + c);
      const v4h s11 = *(const v4h*)(xt + ix.w + c);
      const v4h r = s00 * w4.x + s01 * w4.y + s10 * w4.z + s11 * w4.w;
      *(v4h*)&Blds[bufo + p * 40 + cl] = r;
    }

    // A fragments straight from global, pre-swizzled: one 32B load each.
    const v16h af0 =
        *(const v16h*)(wp + (size_t)s * 8192 + (wv * 2 + 0) * 512 + lane * 16);
    const v16h af1 =
        *(const v16h*)(wp + (size_t)s * 8192 + (wv * 2 + 1) * 512 + lane * 16);
    if (s + 1 < KSTEPS)
      __builtin_prefetch(wp + (size_t)(s + 1) * 8192 + wv * 1024 + lane * 16,
                         0, 0);

    __syncthreads();  // B tile ready (prev buffer's readers already passed)

    // B fragments: lane needs K = hl*8 + 0..7 and 16 + hl*8 + 0..7 for column n.
#pragma unroll
    for (int nt = 0; nt < 4; ++nt) {
      const int n = nt * 16 + l15;
      const _Float16* bp = &Blds[bufo + n * 40 + hl * 8];
      const v8h lo = *(const v8h*)bp;
      const v8h hi = *(const v8h*)(bp + 16);
      const v16h bf = __builtin_shufflevector(lo, hi, 0, 1, 2, 3, 4, 5, 6, 7,
                                              8, 9, 10, 11, 12, 13, 14, 15);
      acc[0][nt] = __builtin_amdgcn_wmma_f32_16x16x32_f16(
          false, af0, false, bf, (short)0, acc[0][nt], false, false);
      acc[1][nt] = __builtin_amdgcn_wmma_f32_16x16x32_f16(
          false, af1, false, bf, (short)0, acc[1][nt], false, false);
    }
  }

  // Epilogue: D reg r, half hl -> M = r + 8*hl; lane l15 -> N. Coalesced f32
  // stores (16 consecutive floats per half-wave), bias fused.
#pragma unroll
  for (int mt = 0; mt < 2; ++mt) {
#pragma unroll
    for (int r = 0; r < 8; ++r) {
      const int cout = wv * 32 + mt * 16 + hl * 8 + r;
      const float bv = bias[cout];
      float* op = out + ((size_t)(b * COUT + cout)) * HW + sp0;
#pragma unroll
      for (int nt = 0; nt < 4; ++nt)
        op[nt * 16 + l15] = acc[mt][nt][r] + bv;
    }
  }
}

// ---------------------------------------------------------------------------
extern "C" void kernel_launch(void* const* d_in, const int* in_sizes, int n_in,
                              void* d_out, int out_size, void* d_ws,
                              size_t ws_size, hipStream_t stream) {
  (void)in_sizes; (void)n_in; (void)out_size; (void)ws_size;
  const float* x    = (const float*)d_in[0];  // (8,256,64,64)
  const float* off  = (const float*)d_in[1];  // (8,18,64,64)
  const float* w    = (const float*)d_in[2];  // (256,256,3,3)
  const float* bias = (const float*)d_in[3];  // (256,)
  float* out = (float*)d_out;                 // (8,256,64,64)

  _Float16* wp = (_Float16*)d_ws;                      // 1.18 MB fragments
  _Float16* xt = (_Float16*)((char*)d_ws + WP_BYTES);  // 16.8 MB NHWC f16

  pack_weight_kernel<<<WP_ELEMS / 256, 256, 0, stream>>>(w, wp);
  transpose_x_kernel<<<dim3(HW / 32, CIN / 32, B_), dim3(32, 8), 0, stream>>>(
      x, xt);
  deform_wmma_kernel<<<(B_ * HW) / NTILE, 256, 0, stream>>>(xt, wp, off, bias,
                                                            out);
}
